// CustomMultiHeadAttention_40183714021840
// MI455X (gfx1250) — matrix operand order repaired and checked by reading, not compiled, not run on hardware
//
#include <hip/hip_runtime.h>
#include <stddef.h>

// ---- problem dims (fixed by the reference) ----
#define N_HEADS_C 16
#define KEY_DIM_C 64
#define D_MODEL_C 1024
#define B_C       2
#define NQ_C      2048
#define NK_C      2048
#define M_ROWS    (B_C * NQ_C)   // 4096 flattened (batch, seq) rows

// ---- CDNA5 WMMA types (wave32, 16x16x32 bf16 -> f32) ----
typedef __attribute__((ext_vector_type(16))) __bf16 v16bf;
typedef __attribute__((ext_vector_type(8)))  float  v8f;
typedef __attribute__((ext_vector_type(4)))  unsigned int u32x4;

union FragU { v16bf v; u32x4 q[2]; };

// Every A/B fragment is exactly two 16-byte LDS loads:
//   A (16x32, 16-bit): lanes 0-15 hold K 0..7 & 16..23, lanes 16-31 hold
//     K 8..15 & 24..31 (pairs per VGPR)  -> p0 = row+kbase, p1 = row+kbase+16
//   B (32x16, 16-bit): lane holds column N=lane%16, K 8g..8g+7 & 8g+16..8g+23
//     (as A; p0 = col+kbase, p1 = +16) from a source stored K-contiguous per column.
__device__ __forceinline__ v16bf frag2(const __bf16* p0, const __bf16* p1) {
  FragU f;
  f.q[0] = *(const u32x4*)p0;
  f.q[1] = *(const u32x4*)p1;
  return f.v;
}

__device__ __forceinline__ v8f v8f_zero() {
  v8f z;
#pragma unroll
  for (int i = 0; i < 8; ++i) z[i] = 0.0f;
  return z;
}

// pack two fp32 -> one dword of two bf16 (compiler folds to cvt_pk/mixlo forms)
__device__ __forceinline__ unsigned int bf16pack(float a, float b) {
  union { __bf16 h; unsigned short u; } x, y;
  x.h = (__bf16)a; y.h = (__bf16)b;
  return (unsigned int)x.u | ((unsigned int)y.u << 16);
}
__device__ __forceinline__ unsigned int bf16pack_raw(__bf16 a, __bf16 b) {
  union { __bf16 h; unsigned short u; } x, y;
  x.h = a; y.h = b;
  return (unsigned int)x.u | ((unsigned int)y.u << 16);
}

#define WMMA_BF16(a, b, c) \
  __builtin_amdgcn_wmma_f32_16x16x32_bf16(false, (a), false, (b), (short)0, (c), false, false)

// ============================================================================
// Kernel 1: fp32 X [4096,1024] @ fp32 W [1024,1024] + bias -> bf16, scattered
// head-major [h*B, n, 64]. 'scale' folds 1/sqrt(dk) into Q.
// Block 256 thr = 8 waves; block tile 64x128; wave tile 32x32 (2x2 WMMA tiles).
// Staging is b128-only on the LDS write side:
//   X tile: 8 fp32 -> packed u32x4 -> 1 ds_store_b128 / thread / step
//   W tile: column-per-thread (16 coalesced b32 loads) -> 2 ds_store_b128
// ============================================================================
__global__ __launch_bounds__(256)
void proj_gemm_bf16(const float* __restrict__ X, const float* __restrict__ W,
                    const float* __restrict__ bias, __bf16* __restrict__ Out,
                    float scale)
{
  __shared__ alignas(16) __bf16 As[64][40];    // X tile  (row-major, 80B stride)
  __shared__ alignas(16) __bf16 Bst[128][40];  // W tile, TRANSPOSED: [n][k]

  const int tid  = threadIdx.x;
  const int lane = tid & 31;
  const int wave = tid >> 5;
  const int ln   = lane & 15;
  const int g    = lane >> 4;
  const int wm   = (wave >> 2) * 32;
  const int wn   = (wave & 3) * 32;
  const int m0   = blockIdx.y * 64;
  const int n0   = blockIdx.x * 128;

  v8f acc[2][2];
#pragma unroll
  for (int i = 0; i < 2; ++i)
#pragma unroll
    for (int j = 0; j < 2; ++j) acc[i][j] = v8f_zero();

  const int arow = tid >> 2;          // 0..63
  const int acol = (tid & 3) * 8;     // 0,8,16,24
  const int bn   = tid & 127;         // W column within tile
  const int bkh  = (tid >> 7) * 16;   // K half: 0 or 16

  for (int k0 = 0; k0 < D_MODEL_C; k0 += 32) {
    {   // X tile 64x32: fp32 -> bf16, one b128 store
      const float* src = X + (size_t)(m0 + arow) * D_MODEL_C + k0 + acol;
      u32x4 p;
#pragma unroll
      for (int i = 0; i < 4; ++i) p[i] = bf16pack(src[2 * i], src[2 * i + 1]);
      *(u32x4*)&As[arow][acol] = p;
    }
    {   // W tile 32x128 transposed: 16 coalesced column loads -> 2 b128 stores
      const float* src = W + (size_t)(k0 + bkh) * D_MODEL_C + n0 + bn;
      float c[16];
#pragma unroll
      for (int i = 0; i < 16; ++i) c[i] = src[(size_t)i * D_MODEL_C];
      u32x4 p0, p1;
#pragma unroll
      for (int i = 0; i < 4; ++i) {
        p0[i] = bf16pack(c[2 * i],     c[2 * i + 1]);
        p1[i] = bf16pack(c[8 + 2 * i], c[9 + 2 * i]);
      }
      *(u32x4*)&Bst[bn][bkh]     = p0;
      *(u32x4*)&Bst[bn][bkh + 8] = p1;
    }
    __syncthreads();

    const int kb = g * 8;
#pragma unroll
    for (int i = 0; i < 2; ++i) {
      const __bf16* ar = &As[wm + i * 16 + ln][kb];
      const v16bf a = frag2(ar, ar + 16);
#pragma unroll
      for (int j = 0; j < 2; ++j) {
        const __bf16* br = &Bst[wn + j * 16 + ln][g * 8];
        acc[i][j] = WMMA_BF16(a, frag2(br, br + 16), acc[i][j]);
      }
    }
    __syncthreads();
  }

  // C layout: VGPR r, lanes 0-15 -> (M=r, N=lane), lanes 16-31 -> (M=r+8, N=lane-16)
#pragma unroll
  for (int i = 0; i < 2; ++i)
#pragma unroll
    for (int j = 0; j < 2; ++j) {
      const int col  = n0 + wn + j * 16 + ln;
      const int head = col >> 6;
      const int d    = col & 63;
      const float bv = bias[col];
#pragma unroll
      for (int r = 0; r < 8; ++r) {
        const int m     = m0 + wm + i * 16 + r + 8 * g;
        const int batch = m >> 11;       // m / NQ
        const int n     = m & 2047;
        const float v   = (acc[i][j][r] + bv) * scale;
        Out[((size_t)(head * B_C + batch) * NQ_C + n) * KEY_DIM_C + d] = (__bf16)v;
      }
    }
}

// ============================================================================
// Kernel 2: flash-style attention per (head*B+batch, 64-query block).
// Block 128 thr = 4 waves; each wave owns 16 query rows, streams 64-key chunks.
// Faithful quirks: multiplicative mask (masked entries contribute exp(0)),
// mask indexed batch-major-repeat (mb = hb / N_HEADS) while rows are head-major.
// V is staged transposed via column-per-thread loads + packed b128 stores.
// ============================================================================
__global__ __launch_bounds__(128)
void attn_flash_bf16(const __bf16* __restrict__ Qd, const __bf16* __restrict__ Kd,
                     const __bf16* __restrict__ Vd, const float* __restrict__ mask,
                     __bf16* __restrict__ Y)
{
  __shared__ alignas(16) __bf16 Ks[64][72];      // K chunk  [key][d]
  __shared__ alignas(16) __bf16 Vt[64][72];      // V chunk transposed [d][key]
  __shared__ alignas(16) __bf16 Ps[4][16][72];   // per-wave P tile (C->A relay)

  const int tid  = threadIdx.x;
  const int lane = tid & 31;
  const int wave = tid >> 5;
  const int ln   = lane & 15;
  const int g    = lane >> 4;
  const int kb   = g * 8;

  const int hb    = blockIdx.x;          // head*B + batch
  const int batch = hb & (B_C - 1);
  const int head  = hb / B_C;
  const int mb    = hb / N_HEADS_C;      // reference's repeat() quirk
  const int q0    = blockIdx.y * 64 + wave * 16;

  // Q fragments for this wave's 16 rows (Q pre-scaled by 1/sqrt(dk))
  const __bf16* qrow = Qd + ((size_t)hb * NQ_C + q0 + ln) * KEY_DIM_C;
  const v16bf aq0 = frag2(qrow + kb,      qrow + kb + 16);
  const v16bf aq1 = frag2(qrow + 32 + kb, qrow + 32 + kb + 16);

  float m_run[8], l_run[8];
#pragma unroll
  for (int r = 0; r < 8; ++r) { m_run[r] = -1e30f; l_run[r] = 0.0f; }
  v8f o[4];
#pragma unroll
  for (int t = 0; t < 4; ++t) o[t] = v8f_zero();

  const int lkey = tid >> 1;            // K loader: row copy
  const int ld0  = (tid & 1) * 32;
  const int vd   = tid & 63;            // V loader: one d-column per thread
  const int vkh  = (tid >> 6) * 32;     // key half: 0 or 32

  for (int key0 = 0; key0 < NK_C; key0 += 64) {
    __syncthreads();                       // all waves done with previous chunk
    {
      // K chunk [64 x 64] row-major: straight b128 copies
      const __bf16* ksrc = Kd + ((size_t)hb * NK_C + key0 + lkey) * KEY_DIM_C + ld0;
      u32x4* kdst = (u32x4*)&Ks[lkey][ld0];
      const u32x4* ksv = (const u32x4*)ksrc;
      kdst[0] = ksv[0]; kdst[1] = ksv[1]; kdst[2] = ksv[2]; kdst[3] = ksv[3];

      // V chunk transposed: 32 coalesced u16 loads down one column, 4 b128 stores
      const __bf16* vsrc = Vd + ((size_t)hb * NK_C + key0 + vkh) * KEY_DIM_C + vd;
      __bf16 c[32];
#pragma unroll
      for (int i = 0; i < 32; ++i) c[i] = vsrc[(size_t)i * KEY_DIM_C];
#pragma unroll
      for (int j = 0; j < 4; ++j) {
        u32x4 p;
#pragma unroll
        for (int i = 0; i < 4; ++i)
          p[i] = bf16pack_raw(c[8 * j + 2 * i], c[8 * j + 2 * i + 1]);
        *(u32x4*)&Vt[vd][vkh + 8 * j] = p;
      }
      if (key0 + 64 < NK_C) {              // global_prefetch_b8 for next chunk
        __builtin_prefetch(Kd + ((size_t)hb * NK_C + key0 + 64 + lkey) * KEY_DIM_C, 0, 0);
        __builtin_prefetch(Vd + ((size_t)hb * NK_C + key0 + 64 + lkey) * KEY_DIM_C, 0, 0);
      }
    }
    __syncthreads();

    // ---- S = Q * K^T  (16x64 per wave, 4 column tiles, K-depth 64 = 2 steps)
    v8f s[4];
#pragma unroll
    for (int t = 0; t < 4; ++t) s[t] = v8f_zero();
#pragma unroll
    for (int kk = 0; kk < 2; ++kk) {
      const v16bf a = kk ? aq1 : aq0;
#pragma unroll
      for (int t = 0; t < 4; ++t) {
        const __bf16* bp = &Ks[t * 16 + ln][kk * 32 + g * 8];
        s[t] = WMMA_BF16(a, frag2(bp, bp + 16), s[t]);
      }
    }

    // ---- multiplicative mask + online softmax (row stats live per 16-lane half)
    float newmax[8];
#pragma unroll
    for (int r = 0; r < 8; ++r) {
      const int q = q0 + r + 8 * g;
      const float* mp = mask + ((size_t)mb * NQ_C + q) * NK_C + key0;
      float mx = -1e30f;
#pragma unroll
      for (int t = 0; t < 4; ++t) {
        const float v = s[t][r] * mp[t * 16 + ln];
        s[t][r] = v;
        mx = fmaxf(mx, v);
      }
      newmax[r] = mx;
    }
#pragma unroll
    for (int x = 1; x < 16; x <<= 1)
#pragma unroll
      for (int r = 0; r < 8; ++r)
        newmax[r] = fmaxf(newmax[r], __shfl_xor(newmax[r], x, 32));

    float alpha[8], lsum[8];
#pragma unroll
    for (int r = 0; r < 8; ++r) {
      const float mnew = fmaxf(m_run[r], newmax[r]);
      alpha[r] = __expf(m_run[r] - mnew);
      m_run[r] = mnew;
      float ls = 0.0f;
#pragma unroll
      for (int t = 0; t < 4; ++t) {
        const float p = __expf(s[t][r] - mnew);
        s[t][r] = p;
        ls += p;
      }
      lsum[r] = ls;
    }
#pragma unroll
    for (int x = 1; x < 16; x <<= 1)
#pragma unroll
      for (int r = 0; r < 8; ++r)
        lsum[r] += __shfl_xor(lsum[r], x, 32);
#pragma unroll
    for (int r = 0; r < 8; ++r) l_run[r] = l_run[r] * alpha[r] + lsum[r];
#pragma unroll
    for (int t = 0; t < 4; ++t)
#pragma unroll
      for (int r = 0; r < 8; ++r) o[t][r] *= alpha[r];

    // ---- stage P (C-layout) into per-wave LDS tile, re-read in A-layout.
    // (This cross-lane relayout is the one unavoidable 16-bit scatter.)
#pragma unroll
    for (int t = 0; t < 4; ++t)
#pragma unroll
      for (int r = 0; r < 8; ++r)
        Ps[wave][r + 8 * g][t * 16 + ln] = (__bf16)s[t][r];
    asm volatile("s_wait_dscnt 0" ::: "memory");   // same-wave LDS RAW fence

    // ---- O += P * V
#pragma unroll
    for (int kk = 0; kk < 2; ++kk) {
      const __bf16* pr = &Ps[wave][ln][kk * 32 + kb];
      const v16bf ap = frag2(pr, pr + 16);
#pragma unroll
      for (int t = 0; t < 4; ++t) {
        const __bf16* bp = &Vt[t * 16 + ln][kk * 32 + g * 8];
        o[t] = WMMA_BF16(ap, frag2(bp, bp + 16), o[t]);
      }
    }
  }

  // ---- normalize, scatter to Y[b, q, head*64 + d] (bf16 for the final GEMM)
#pragma unroll
  for (int r = 0; r < 8; ++r) {
    const float inv = 1.0f / l_run[r];
    const int q = q0 + r + 8 * g;
#pragma unroll
    for (int t = 0; t < 4; ++t) {
      const int d = t * 16 + ln;
      Y[((size_t)batch * NQ_C + q) * D_MODEL_C + head * KEY_DIM_C + d] =
          (__bf16)(o[t][r] * inv);
    }
  }
}

// ============================================================================
// Kernel 3: Y(bf16) [4096,1024] @ Wo(fp32->bf16) + bo -> fp32 d_out.
// ============================================================================
__global__ __launch_bounds__(256)
void out_gemm_bf16(const __bf16* __restrict__ A, const float* __restrict__ W,
                   const float* __restrict__ bias, float* __restrict__ C)
{
  __shared__ alignas(16) __bf16 As[64][40];
  __shared__ alignas(16) __bf16 Bst[128][40];

  const int tid  = threadIdx.x;
  const int lane = tid & 31;
  const int wave = tid >> 5;
  const int ln   = lane & 15;
  const int g    = lane >> 4;
  const int wm   = (wave >> 2) * 32;
  const int wn   = (wave & 3) * 32;
  const int m0   = blockIdx.y * 64;
  const int n0   = blockIdx.x * 128;

  v8f acc[2][2];
#pragma unroll
  for (int i = 0; i < 2; ++i)
#pragma unroll
    for (int j = 0; j < 2; ++j) acc[i][j] = v8f_zero();

  const int arow = tid >> 2;
  const int acol = (tid & 3) * 8;
  const int bn   = tid & 127;
  const int bkh  = (tid >> 7) * 16;

  for (int k0 = 0; k0 < D_MODEL_C; k0 += 32) {
    {   // Y tile already bf16: straight 16B copy
      const __bf16* src = A + (size_t)(m0 + arow) * D_MODEL_C + k0 + acol;
      *(u32x4*)&As[arow][acol] = *(const u32x4*)src;
    }
    {   // Wo tile transposed: column loads + packed b128 stores
      const float* src = W + (size_t)(k0 + bkh) * D_MODEL_C + n0 + bn;
      float c[16];
#pragma unroll
      for (int i = 0; i < 16; ++i) c[i] = src[(size_t)i * D_MODEL_C];
      u32x4 p0, p1;
#pragma unroll
      for (int i = 0; i < 4; ++i) {
        p0[i] = bf16pack(c[2 * i],     c[2 * i + 1]);
        p1[i] = bf16pack(c[8 + 2 * i], c[9 + 2 * i]);
      }
      *(u32x4*)&Bst[bn][bkh]     = p0;
      *(u32x4*)&Bst[bn][bkh + 8] = p1;
    }
    __syncthreads();

    const int kb = g * 8;
#pragma unroll
    for (int i = 0; i < 2; ++i) {
      const __bf16* ar = &As[wm + i * 16 + ln][kb];
      const v16bf a = frag2(ar, ar + 16);
#pragma unroll
      for (int j = 0; j < 2; ++j) {
        const __bf16* br = &Bst[wn + j * 16 + ln][g * 8];
        acc[i][j] = WMMA_BF16(a, frag2(br, br + 16), acc[i][j]);
      }
    }
    __syncthreads();
  }

#pragma unroll
  for (int i = 0; i < 2; ++i)
#pragma unroll
    for (int j = 0; j < 2; ++j) {
      const int col  = n0 + wn + j * 16 + ln;
      const float bv = bias[col];
#pragma unroll
      for (int r = 0; r < 8; ++r) {
        const int m = m0 + wm + i * 16 + r + 8 * g;
        C[(size_t)m * D_MODEL_C + col] = acc[i][j][r] + bv;
      }
    }
}

// ============================================================================
extern "C" void kernel_launch(void* const* d_in, const int* in_sizes, int n_in,
                              void* d_out, int out_size, void* d_ws, size_t ws_size,
                              hipStream_t stream)
{
  (void)in_sizes; (void)n_in; (void)out_size; (void)ws_size;

  // setup_inputs() order: Q, V, K, mask, Wq, bq, Wk, bk, Wv, bv, Wo, bo
  const float* Q    = (const float*)d_in[0];
  const float* V    = (const float*)d_in[1];
  const float* K    = (const float*)d_in[2];
  const float* mask = (const float*)d_in[3];
  const float* Wq   = (const float*)d_in[4];
  const float* bq   = (const float*)d_in[5];
  const float* Wk   = (const float*)d_in[6];
  const float* bk   = (const float*)d_in[7];
  const float* Wv   = (const float*)d_in[8];
  const float* bv   = (const float*)d_in[9];
  const float* Wo   = (const float*)d_in[10];
  const float* bo   = (const float*)d_in[11];

  // Workspace: Qd/Kd/Vd head-major bf16 + Y bf16 = 4 x 8 MB = 32 MB
  const size_t projElems = (size_t)N_HEADS_C * B_C * NK_C * KEY_DIM_C;
  __bf16* Qd = (__bf16*)d_ws;
  __bf16* Kd = Qd + projElems;
  __bf16* Vd = Kd + projElems;
  __bf16* Yb = Vd + projElems;

  dim3 gGemm(D_MODEL_C / 128, M_ROWS / 64);   // (8, 64)
  proj_gemm_bf16<<<gGemm, 256, 0, stream>>>(Q, Wq, bq, Qd, 0.125f); // 1/sqrt(64)
  proj_gemm_bf16<<<gGemm, 256, 0, stream>>>(K, Wk, bk, Kd, 1.0f);
  proj_gemm_bf16<<<gGemm, 256, 0, stream>>>(V, Wv, bv, Vd, 1.0f);

  dim3 gAttn(N_HEADS_C * B_C, NQ_C / 64);     // (32, 32)
  attn_flash_bf16<<<gAttn, 128, 0, stream>>>(Qd, Kd, Vd, mask, Yb);

  out_gemm_bf16<<<gGemm, 256, 0, stream>>>(Yb, Wo, bo, (float*)d_out);
}
